// JK_GAMLP_17867063951760
// MI455X (gfx1250) — compile-verified, matrix operand order checked by reading, not run on hardware
//
#include <hip/hip_runtime.h>

typedef unsigned short ushort_t;
typedef __attribute__((ext_vector_type(16))) __bf16 v16bf;
typedef __attribute__((ext_vector_type(8)))  float  v8f;
typedef __attribute__((ext_vector_type(4)))  unsigned int v4u;

#define HOPS   8
#define FDIM   128
#define HID    256
#define NCLS   64
#define CONCATK (HOPS*FDIM)   // 1024

__device__ __forceinline__ ushort_t f2bf(float x) {
    union { float f; unsigned u; } v; v.f = x;
    unsigned r = v.u + 0x7FFFu + ((v.u >> 16) & 1u);  // round-to-nearest-even
    return (ushort_t)(r >> 16);
}
__device__ __forceinline__ float bf2f(ushort_t x) {
    union { float f; unsigned u; } v; v.u = ((unsigned)x) << 16;
    return v.f;
}

union Frag {
    v16bf    v;
    ushort_t u[16];
    v4u      q[2];
};

// ---- Pack fp32 weight [K][N] into bf16 WMMA B-fragment order --------------
// Layout: for col-tile ct, k-step ks, lane l: 16 contiguous bf16 values
// (lane's 16 halves of the 32x16 bf16 B operand). Lane l covers n = ct*16 +
// (l&15); K pattern: kb = ks*32 + (l>=16 ? 8 : 0), half j -> k = kb + (j<8 ? j : j+8).
__global__ void pack_weights(const float* __restrict__ W, ushort_t* __restrict__ out,
                             int K, int N, int total) {
    int t = blockIdx.x * blockDim.x + threadIdx.x;
    if (t >= total) return;
    int j    = t & 15;
    int lane = (t >> 4) & 31;
    int rest = t >> 9;
    int ksteps = K >> 5;
    int ks = rest % ksteps;
    int ct = rest / ksteps;
    int n = ct * 16 + (lane & 15);
    int k = ks * 32 + ((lane >> 4) & 1) * 8 + (j < 8 ? j : j + 8);
    out[t] = f2bf(W[(size_t)k * N + n]);
}

// ---- Fragment loaders -----------------------------------------------------
// A (16xK bf16, row-major in LDS, 16 rows, row stride `stride`):
__device__ __forceinline__ void loadA_lds(Frag& fr, const ushort_t* base, int stride,
                                          int ks, int lane) {
    const int m  = lane & 15;
    const int kb = ks * 32 + ((lane >> 4) & 1) * 8;
    const ushort_t* p = base + m * stride + kb;
#pragma unroll
    for (int j = 0; j < 8; ++j)  fr.u[j] = p[j];
#pragma unroll
    for (int j = 8; j < 16; ++j) fr.u[j] = p[j + 8];
}
// B from pre-packed global weights (two 16B loads per lane, contiguous per lane):
__device__ __forceinline__ void loadB_pack(Frag& fr, const ushort_t* wp, int K,
                                           int ct, int ks, int lane) {
    const v4u* p = (const v4u*)(wp + (((size_t)(ct * (K >> 5) + ks)) * 32 + lane) * 16);
    fr.q[0] = p[0];
    fr.q[1] = p[1];
}

// ---- 16-row x 64-col GEMM slice for one wave: D = PReLU(A*W + bias) -> LDS bf16
template <int K>
__device__ __forceinline__ void gemm_tile4(const ushort_t* Asrc, int astride,
                                           const ushort_t* Wp, const float* bias,
                                           float alpha, ushort_t* Dst, int dstride,
                                           int w, int lane) {
    v8f acc[4];
#pragma unroll
    for (int c = 0; c < 4; ++c) {
        float b = bias[(w * 4 + c) * 16 + (lane & 15)];
#pragma unroll
        for (int i = 0; i < 8; ++i) acc[c][i] = b;
    }
    for (int ks = 0; ks < (K >> 5); ++ks) {
        Frag A;
        loadA_lds(A, Asrc, astride, ks, lane);
#pragma unroll
        for (int c = 0; c < 4; ++c) {
            Frag B;
            loadB_pack(B, Wp, K, w * 4 + c, ks, lane);
            acc[c] = __builtin_amdgcn_wmma_f32_16x16x32_bf16(
                false, A.v, false, B.v, (short)0, acc[c], false, false);
        }
    }
#pragma unroll
    for (int c = 0; c < 4; ++c) {
        int col = (w * 4 + c) * 16 + (lane & 15);
#pragma unroll
        for (int i = 0; i < 8; ++i) {
            int m = i + ((lane >> 4) & 1) * 8;
            float x = acc[c][i];
            x = (x >= 0.f) ? x : alpha * x;
            Dst[m * dstride + col] = f2bf(x);
        }
    }
}

// ---- Fused JK-GAMLP kernel: one block (4 waves) = one 16-node tile --------
__launch_bounds__(128)
__global__ void gamlp_fused(const float* __restrict__ feats,
                            const ushort_t* __restrict__ Wjk1p, const float* __restrict__ bjk1,
                            const ushort_t* __restrict__ Wjk2p, const float* __restrict__ bjk2,
                            const float* a_jk_p, const float* a_main_p, const float* a_out_p,
                            const float* __restrict__ w_att_ref,
                            const float* __restrict__ w_att_x, const float* b_att_p,
                            const ushort_t* __restrict__ Wo1p, const float* __restrict__ bo1,
                            const ushort_t* __restrict__ Wo2p, const float* __restrict__ bo2,
                            float* __restrict__ out, int Nnodes) {
    __shared__ ushort_t sFeat[16 * CONCATK];   // concat layout: [row][hop*F+f], bf16
    __shared__ ushort_t sH[16 * HID];          // hidden activations, bf16
    __shared__ ushort_t sJK[16 * HID];         // jk_ref, bf16
    __shared__ ushort_t sAgg[16 * FDIM];       // aggregated features, bf16
    __shared__ float    sSx[16][HOPS];
    __shared__ float    sSref[16];
    __shared__ float    sWsm[16][HOPS];

    const int t    = threadIdx.x;
    const int lane = t & 31;
    const int w    = t >> 5;
    const int n0   = blockIdx.x * 16;

    // ---- Stage feats tile -> LDS bf16 (single HBM read of feats) ----------
    for (int idx = t; idx < 16 * (CONCATK / 4); idx += 128) {
        int r  = idx / (CONCATK / 4);
        int c4 = idx % (CONCATK / 4);
        int h  = (c4 * 4) / FDIM;
        int f  = (c4 * 4) % FDIM;
        int nn = n0 + r; if (nn >= Nnodes) nn = Nnodes - 1;
        const float4 vv = *(const float4*)(feats + ((size_t)h * Nnodes + nn) * FDIM + f);
        ushort_t* d = &sFeat[r * CONCATK + c4 * 4];
        d[0] = f2bf(vv.x); d[1] = f2bf(vv.y); d[2] = f2bf(vv.z); d[3] = f2bf(vv.w);
    }
    __syncthreads();

    // ---- Per-hop attention dots s_x[m][h] (128 threads = 16 rows x 8 hops) -
    {
        int m = t & 15, h = t >> 4;
        float s = 0.f;
        const ushort_t* row = &sFeat[m * CONCATK + h * FDIM];
        for (int f = 0; f < FDIM; ++f) s += bf2f(row[f]) * w_att_x[f];
        sSx[m][h] = s;
    }

    // ---- GEMM1: concat[16x1024] @ W_jk1 -> PReLU(a_jk) -> sH --------------
    gemm_tile4<CONCATK>(sFeat, CONCATK, Wjk1p, bjk1, *a_jk_p, sH, HID, w, lane);
    __syncthreads();

    // ---- GEMM2: sH[16x256] @ W_jk2 -> PReLU(a_main) -> sJK ----------------
    gemm_tile4<HID>(sH, HID, Wjk2p, bjk2, *a_main_p, sJK, HID, w, lane);
    __syncthreads();

    // ---- s_ref + sigmoid + softmax over hops (threads 0..15, one row each)
    if (t < 16) {
        float sr = 0.f;
        const ushort_t* row = &sJK[t * HID];
        for (int c = 0; c < HID; ++c) sr += bf2f(row[c]) * w_att_ref[c];
        sSref[t] = sr;
        float batt = *b_att_p;
        float sc[HOPS], mx = -1e30f;
#pragma unroll
        for (int h = 0; h < HOPS; ++h) {
            float z = sSx[t][h] + sr + batt;
            sc[h] = 1.f / (1.f + __expf(-z));     // sigmoid
            mx = fmaxf(mx, sc[h]);
        }
        float den = 0.f;
#pragma unroll
        for (int h = 0; h < HOPS; ++h) { sc[h] = __expf(sc[h] - mx); den += sc[h]; }
        float inv = 1.f / den;
#pragma unroll
        for (int h = 0; h < HOPS; ++h) sWsm[t][h] = sc[h] * inv;
    }
    __syncthreads();

    // ---- Weighted hop aggregation: agg[m][f] = sum_h W[m][h]*feat[h][m][f] -
    {
        int m = t >> 3;
        int fb = (t & 7) * 16;
        float wv[HOPS];
#pragma unroll
        for (int h = 0; h < HOPS; ++h) wv[h] = sWsm[m][h];
#pragma unroll
        for (int f = fb; f < fb + 16; ++f) {
            float s = 0.f;
#pragma unroll
            for (int h = 0; h < HOPS; ++h)
                s += wv[h] * bf2f(sFeat[m * CONCATK + h * FDIM + f]);
            sAgg[m * FDIM + f] = f2bf(s);
        }
    }
    __syncthreads();

    // ---- GEMM3: agg[16x128] @ W_o1 -> PReLU(a_out) -> sH (reuse) ----------
    gemm_tile4<FDIM>(sAgg, FDIM, Wo1p, bo1, *a_out_p, sH, HID, w, lane);
    __syncthreads();

    // ---- GEMM4: sH[16x256] @ W_o2[256x64] + b_o2 -> global out (f32) ------
    {
        v8f acc;
        const int ct = w;                       // 4 waves cover NCLS=64
        float b = bo2[ct * 16 + (lane & 15)];
#pragma unroll
        for (int i = 0; i < 8; ++i) acc[i] = b;
        for (int ks = 0; ks < (HID >> 5); ++ks) {
            Frag A; loadA_lds(A, sH, HID, ks, lane);
            Frag B; loadB_pack(B, Wo2p, HID, ct, ks, lane);
            acc = __builtin_amdgcn_wmma_f32_16x16x32_bf16(
                false, A.v, false, B.v, (short)0, acc, false, false);
        }
        int col = ct * 16 + (lane & 15);
#pragma unroll
        for (int i = 0; i < 8; ++i) {
            int m = i + ((lane >> 4) & 1) * 8;
            int n = n0 + m;
            if (n < Nnodes) out[(size_t)n * NCLS + col] = acc[i];
        }
    }
}

extern "C" void kernel_launch(void* const* d_in, const int* in_sizes, int n_in,
                              void* d_out, int out_size, void* d_ws, size_t ws_size,
                              hipStream_t stream) {
    (void)n_in; (void)out_size; (void)ws_size;
    const float* feats     = (const float*)d_in[0];
    const float* W_jk1     = (const float*)d_in[1];
    const float* b_jk1     = (const float*)d_in[2];
    const float* W_jk2     = (const float*)d_in[3];
    const float* b_jk2     = (const float*)d_in[4];
    const float* a_jk      = (const float*)d_in[5];
    const float* a_main    = (const float*)d_in[6];
    const float* a_out     = (const float*)d_in[7];
    const float* w_att_ref = (const float*)d_in[8];
    const float* w_att_x   = (const float*)d_in[9];
    const float* b_att     = (const float*)d_in[10];
    const float* W_o1      = (const float*)d_in[11];
    const float* b_o1      = (const float*)d_in[12];
    const float* W_o2      = (const float*)d_in[13];
    const float* b_o2      = (const float*)d_in[14];
    float* out = (float*)d_out;

    const int N = in_sizes[0] / (HOPS * FDIM);

    // bf16 fragment-packed weights in workspace
    ushort_t* ws    = (ushort_t*)d_ws;
    ushort_t* Wjk1p = ws;                                     // 1024*256
    ushort_t* Wjk2p = Wjk1p + CONCATK * HID;                  //  256*256
    ushort_t* Wo1p  = Wjk2p + HID * HID;                      //  128*256
    ushort_t* Wo2p  = Wo1p  + FDIM * HID;                     //  256*64

    pack_weights<<<(CONCATK * HID) / 256, 256, 0, stream>>>(W_jk1, Wjk1p, CONCATK, HID, CONCATK * HID);
    pack_weights<<<(HID * HID) / 256, 256, 0, stream>>>(W_jk2, Wjk2p, HID, HID, HID * HID);
    pack_weights<<<(FDIM * HID) / 256, 256, 0, stream>>>(W_o1, Wo1p, FDIM, HID, FDIM * HID);
    pack_weights<<<(HID * NCLS) / 256, 256, 0, stream>>>(W_o2, Wo2p, HID, NCLS, HID * NCLS);

    const int ntiles = (N + 15) / 16;
    gamlp_fused<<<ntiles, 128, 0, stream>>>(feats,
                                            Wjk1p, b_jk1, Wjk2p, b_jk2,
                                            a_jk, a_main, a_out,
                                            w_att_ref, w_att_x, b_att,
                                            Wo1p, b_o1, Wo2p, b_o2,
                                            out, N);
}
